// LSTMTagger_CRF_53188874994354
// MI455X (gfx1250) — matrix operand order repaired and checked by reading, not compiled, not run on hardware
//
#include <hip/hip_runtime.h>
#include <math.h>

#define Bn     128
#define Tn     512
#define Dn     512
#define NTAGS  52
#define NPAD   64
#define NLAB   32
#define START  (NTAGS - 2)   // 50
#define STOP   (NTAGS - 1)   // 51
#define KCH    (Dn / 32)     // 16 K-chunks of 32

typedef __attribute__((ext_vector_type(16))) _Float16 v16h;
typedef __attribute__((ext_vector_type(8)))  float    v8f;

// Pack 16 consecutive f32 (as four float4) into a v16h fragment.
__device__ __forceinline__ v16h cvt16(float4 f0, float4 f1, float4 f2, float4 f3) {
    v16h r;
    r[0]  = (_Float16)f0.x; r[1]  = (_Float16)f0.y; r[2]  = (_Float16)f0.z; r[3]  = (_Float16)f0.w;
    r[4]  = (_Float16)f1.x; r[5]  = (_Float16)f1.y; r[6]  = (_Float16)f1.z; r[7]  = (_Float16)f1.w;
    r[8]  = (_Float16)f2.x; r[9]  = (_Float16)f2.y; r[10] = (_Float16)f2.z; r[11] = (_Float16)f2.w;
    r[12] = (_Float16)f3.x; r[13] = (_Float16)f3.y; r[14] = (_Float16)f3.z; r[15] = (_Float16)f3.w;
    return r;
}

// -------------------------------------------------------------------------
// Kernel 1: tag_space = lstm_out @ W_tag^T + b_tag via v_wmma_f32_16x16x32_f16.
// W_tag is pre-converted to f16 WMMA B-fragments in LDS (zero-padded to 64
// cols) once per block; each wave then computes two 16-row M tiles x 64 cols.
// -------------------------------------------------------------------------
__global__ __launch_bounds__(128) void tag_space_wmma_kernel(
    const float* __restrict__ lstm_out,   // (Bn*Tn, Dn) row-major
    const float* __restrict__ W_tag,      // (NTAGS, Dn) row-major
    const float* __restrict__ b_tag,      // (NTAGS)
    float* __restrict__ tag_space)        // (Bn*Tn, NPAD)
{
    // B-fragment store: [kc][ntile][lane] -> v16h (32B). 16*4*32*32B = 64 KB.
    __shared__ v16h Bfrags[KCH * 4 * 32];

    const int tid  = threadIdx.x;
    const int wave = tid >> 5;
    const int lane = tid & 31;

    // ---- fill phase: convert W_tag to B fragments (B layout: N = lane&15,
    // lanes 0-15 hold K 0..15 of the chunk, lanes 16-31 hold K 16..31) ----
    for (int s = tid; s < KCH * 4 * 32; s += 128) {
        const int kc = s >> 7;           // /128
        const int nt = (s >> 5) & 3;
        const int ln = s & 31;
        const int col = nt * 16 + (ln & 15);
        const int kb  = (ln < 16) ? 0 : 16;
        v16h frag = {};
        if (col < NTAGS) {
            const float4* pb = (const float4*)(W_tag + (long)col * Dn + kc * 32 + kb);
            frag = cvt16(pb[0], pb[1], pb[2], pb[3]);
        }
        Bfrags[s] = frag;
    }
    __syncthreads();

    // ---- compute phase: two M tiles per wave ----
    const long tile0 = ((long)blockIdx.x * 4 + wave) * 2;
    const long mbase = tile0 * 16;                       // 32 rows per wave

    const int mrow = lane & 15;
    // A layout: lanes 0-15 hold K 0..15, lanes 16-31 K 16..31.
    const int ka = (lane < 16) ? 0 : 16;
    const float* arow0 = lstm_out + (mbase + mrow) * Dn;
    const float* arow1 = arow0 + 16 * Dn;

    v8f accA0 = {}, accA1 = {}, accA2 = {}, accA3 = {};
    v8f accB0 = {}, accB1 = {}, accB2 = {}, accB3 = {};

    for (int kc = 0; kc < KCH; ++kc) {
        const float4* pa0 = (const float4*)(arow0 + kc * 32 + ka);
        const float4* pa1 = (const float4*)(arow1 + kc * 32 + ka);
        const v16h a0 = cvt16(pa0[0], pa0[1], pa0[2], pa0[3]);  // +0,+4,+8,+12
        const v16h a1 = cvt16(pa1[0], pa1[1], pa1[2], pa1[3]);

        const v16h* bf = &Bfrags[(kc * 4) * 32 + lane];
        const v16h b0 = bf[0];
        const v16h b1 = bf[32];
        const v16h b2 = bf[64];
        const v16h b3 = bf[96];

        accA0 = __builtin_amdgcn_wmma_f32_16x16x32_f16(false, a0, false, b0, (short)0, accA0, false, false);
        accB0 = __builtin_amdgcn_wmma_f32_16x16x32_f16(false, a1, false, b0, (short)0, accB0, false, false);
        accA1 = __builtin_amdgcn_wmma_f32_16x16x32_f16(false, a0, false, b1, (short)0, accA1, false, false);
        accB1 = __builtin_amdgcn_wmma_f32_16x16x32_f16(false, a1, false, b1, (short)0, accB1, false, false);
        accA2 = __builtin_amdgcn_wmma_f32_16x16x32_f16(false, a0, false, b2, (short)0, accA2, false, false);
        accB2 = __builtin_amdgcn_wmma_f32_16x16x32_f16(false, a1, false, b2, (short)0, accB2, false, false);
        accA3 = __builtin_amdgcn_wmma_f32_16x16x32_f16(false, a0, false, b3, (short)0, accA3, false, false);
        accB3 = __builtin_amdgcn_wmma_f32_16x16x32_f16(false, a1, false, b3, (short)0, accB3, false, false);
    }

    // C layout: element r of lane = (M = r + (lane<16?0:8), N = lane&15)
    const int moff = (lane < 16) ? 0 : 8;
    const int ncol = lane & 15;
    #pragma unroll
    for (int j = 0; j < 4; ++j) {
        const int col = j * 16 + ncol;
        const float bt = (col < NTAGS) ? b_tag[col] : 0.0f;
        const v8f aA = (j == 0) ? accA0 : (j == 1) ? accA1 : (j == 2) ? accA2 : accA3;
        const v8f aB = (j == 0) ? accB0 : (j == 1) ? accB1 : (j == 2) ? accB2 : accB3;
        #pragma unroll
        for (int r = 0; r < 8; ++r) {
            tag_space[(mbase + moff + r) * NPAD + col]      = aA[r] + bt;
            tag_space[(mbase + 16 + moff + r) * NPAD + col] = aB[r] + bt;
        }
    }
}

// -------------------------------------------------------------------------
// Kernel 2: pooled max over T (L2-resident re-read), intent logits,
// log_softmax + argmax via wave32 shuffles. One block per batch row.
// -------------------------------------------------------------------------
__global__ __launch_bounds__(256) void intent_kernel(
    const float* __restrict__ lstm_out,
    const float* __restrict__ W_int,      // (NLAB, Dn)
    const float* __restrict__ b_int,      // (NLAB)
    float* __restrict__ out_labels,       // (Bn)
    float* __restrict__ out_lsm)          // (Bn, NLAB)
{
    const int b = blockIdx.x;
    __shared__ float pooled[Dn];

    for (int d = threadIdx.x; d < Dn; d += 256) {
        const float* p = lstm_out + ((long)b * Tn) * Dn + d;
        float mx = -1e30f;
        for (int t = 0; t < Tn; ++t) {
            if (t + 8 < Tn) __builtin_prefetch(p + (long)(t + 8) * Dn, 0, 1);
            mx = fmaxf(mx, p[(long)t * Dn]);
        }
        pooled[d] = mx;
    }
    __syncthreads();

    if (threadIdx.x < NLAB) {
        const int l = threadIdx.x;
        float acc = b_int[l];
        const float* wrow = W_int + (long)l * Dn;
        for (int k = 0; k < Dn; ++k) acc += pooled[k] * wrow[k];

        float mx = acc;
        #pragma unroll
        for (int o = 16; o > 0; o >>= 1) mx = fmaxf(mx, __shfl_xor(mx, o, 32));
        float s = expf(acc - mx);
        #pragma unroll
        for (int o = 16; o > 0; o >>= 1) s += __shfl_xor(s, o, 32);
        const float lsm = acc - mx - logf(s);
        out_lsm[(long)b * NLAB + l] = lsm;

        float v = lsm; int idx = l;
        #pragma unroll
        for (int o = 16; o > 0; o >>= 1) {
            float v2 = __shfl_xor(v, o, 32);
            int   i2 = __shfl_xor(idx, o, 32);
            if (v2 > v || (v2 == v && i2 < idx)) { v = v2; idx = i2; }
        }
        if (l == 0) out_labels[b] = (float)idx;
    }
}

// -------------------------------------------------------------------------
// Kernel 3: Viterbi forward scan. One block (64 threads) per batch row.
// Thread j keeps transition column j in 52 VGPRs (fully unrolled), reads
// only broadcast v[i] from LDS; double-buffered v => 1 barrier / step.
// -------------------------------------------------------------------------
__global__ __launch_bounds__(64) void viterbi_fwd_kernel(
    const float* __restrict__ tag_space,          // (Bn*Tn, NPAD)
    const unsigned char* __restrict__ masks,      // (Bn, Tn) bool
    const float* __restrict__ trans,              // (NTAGS, NTAGS)
    float* __restrict__ v_fin,                    // (Bn, NPAD)
    unsigned char* __restrict__ bp)               // (Bn, Tn-1, NPAD)
{
    const int b = blockIdx.x;
    const int j = threadIdx.x;

    __shared__ float vbuf[2][NPAD];

    float tcol[NTAGS];
    if (j < NTAGS) {
        #pragma unroll
        for (int i = 0; i < NTAGS; ++i) tcol[i] = trans[i * NTAGS + j];
        vbuf[0][j] = trans[START * NTAGS + j] + tag_space[((long)b * Tn) * NPAD + j];
    }
    __syncthreads();

    int p = 0;
    for (int t = 1; t < Tn; ++t) {
        const unsigned char m = masks[(long)b * Tn + t];
        if (j < NTAGS) {
            const float f = tag_space[((long)b * Tn + t) * NPAD + j];
            float best = -1e30f; int bi = 0;
            #pragma unroll
            for (int i = 0; i < NTAGS; ++i) {
                const float s = vbuf[p][i] + tcol[i];
                if (s > best) { best = s; bi = i; }   // strict > => first max wins
            }
            const long bpo = ((long)b * (Tn - 1) + (t - 1)) * NPAD + j;
            if (m) { vbuf[1 ^ p][j] = best + f;     bp[bpo] = (unsigned char)bi; }
            else   { vbuf[1 ^ p][j] = vbuf[p][j];   bp[bpo] = (unsigned char)j;  }
        }
        __syncthreads();
        p ^= 1;
    }
    if (j < NTAGS) v_fin[(long)b * NPAD + j] = vbuf[p][j];
}

// -------------------------------------------------------------------------
// Kernel 4: terminal score + backtrace. One wave32 per batch row.
// Lanes stage 32 bp rows (2 KB) per chunk into LDS (prefetching the next
// chunk), so the serial walk is LDS byte reads, not dependent global loads.
// -------------------------------------------------------------------------
__global__ __launch_bounds__(32) void viterbi_back_kernel(
    const float* __restrict__ v_fin,
    const float* __restrict__ trans,
    const unsigned char* __restrict__ bp,
    float* __restrict__ out_path,     // (Bn)
    float* __restrict__ out_best)     // (Bn, Tn)
{
    const int b    = blockIdx.x;
    const int lane = threadIdx.x;

    __shared__ float4 buf[32 * 4];    // 32 rows x 64B

    // terminal argmax (tie -> lowest index, matches jnp.argmax)
    float best = -1e30f; int tag = -1;
    for (int jj = lane; jj < NTAGS; jj += 32) {
        const float s = v_fin[(long)b * NPAD + jj] + trans[jj * NTAGS + STOP];
        if (s > best) { best = s; tag = jj; }
    }
    #pragma unroll
    for (int o = 16; o > 0; o >>= 1) {
        float s2 = __shfl_xor(best, o, 32);
        int   t2 = __shfl_xor(tag, o, 32);
        if (s2 > best || (s2 == best && t2 >= 0 && (tag < 0 || t2 < tag))) { best = s2; tag = t2; }
    }
    if (lane == 0) {
        out_path[b] = best;
        out_best[(long)b * Tn + (Tn - 1)] = (float)tag;
    }

    const float4* bpv = (const float4*)(bp + (long)b * (Tn - 1) * NPAD);

    // prologue: load chunk 15 (rows 480..510)
    float4 r0 = {}, r1 = {}, r2 = {}, r3 = {};
    {
        const int row = 15 * 32 + lane;
        if (row < Tn - 1) {
            r0 = bpv[row * 4 + 0]; r1 = bpv[row * 4 + 1];
            r2 = bpv[row * 4 + 2]; r3 = bpv[row * 4 + 3];
        }
    }

    for (int c = 15; c >= 0; --c) {
        buf[lane * 4 + 0] = r0; buf[lane * 4 + 1] = r1;
        buf[lane * 4 + 2] = r2; buf[lane * 4 + 3] = r3;
        __syncthreads();                       // single-wave: near-free
        if (c > 0) {                           // prefetch next chunk
            const int row = (c - 1) * 32 + lane;
            r0 = bpv[row * 4 + 0]; r1 = bpv[row * 4 + 1];
            r2 = bpv[row * 4 + 2]; r3 = bpv[row * 4 + 3];
        }
        if (lane == 0) {
            const unsigned char* lb = (const unsigned char*)buf;
            const int lo = c * 32;
            const int hi = (c == 15) ? (Tn - 2) : (lo + 31);
            for (int r = hi; r >= lo; --r) {
                tag = lb[(r - lo) * NPAD + tag];
                out_best[(long)b * Tn + r] = (float)tag;
            }
        }
        __syncthreads();
        tag = __shfl(tag, 0, 32);              // keep all lanes' tag coherent
    }
}

// -------------------------------------------------------------------------
extern "C" void kernel_launch(void* const* d_in, const int* in_sizes, int n_in,
                              void* d_out, int out_size, void* d_ws, size_t ws_size,
                              hipStream_t stream) {
    const float*         lstm_out = (const float*)d_in[0];
    const unsigned char* masks    = (const unsigned char*)d_in[1];
    const float*         W_tag    = (const float*)d_in[2];
    const float*         b_tag    = (const float*)d_in[3];
    const float*         trans    = (const float*)d_in[4];
    const float*         W_int    = (const float*)d_in[5];
    const float*         b_int    = (const float*)d_in[6];

    float* out        = (float*)d_out;
    float* out_path   = out;                          // (Bn)
    float* out_best   = out + Bn;                     // (Bn, Tn)
    float* out_labels = out + Bn + Bn * Tn;           // (Bn)
    float* out_lsm    = out + Bn + Bn * Tn + Bn;      // (Bn, NLAB)

    char* ws = (char*)d_ws;
    float*         tag_space = (float*)ws;                                   // Bn*Tn*NPAD f32
    unsigned char* bp        = (unsigned char*)(ws + (size_t)Bn * Tn * NPAD * 4);
    float*         v_fin     = (float*)(ws + (size_t)Bn * Tn * NPAD * 4
                                           + (size_t)Bn * (Tn - 1) * NPAD);  // Bn*NPAD f32

    const int mtiles = (Bn * Tn) / 16;                   // 4096
    tag_space_wmma_kernel<<<mtiles / 8, 128, 0, stream>>>(lstm_out, W_tag, b_tag, tag_space);
    intent_kernel<<<Bn, 256, 0, stream>>>(lstm_out, W_int, b_int, out_labels, out_lsm);
    viterbi_fwd_kernel<<<Bn, 64, 0, stream>>>(tag_space, masks, trans, v_fin, bp);
    viterbi_back_kernel<<<Bn, 32, 0, stream>>>(v_fin, trans, bp, out_path, out_best);
}
